// LSTMCell_7078106104555
// MI455X (gfx1250) — matrix operand order repaired and checked by reading, not compiled, not run on hardware
//
#include <hip/hip_runtime.h>
#include <hip/hip_bf16.h>

#define SEQ  512
#define BB   64
#define NIN  1024
#define NHID 1024
#define G4   4096            // 4 * NHID
#define NBLK 80
#define TPB  256             // 8 wave32 waves per block
#define WPB  (TPB / 32)      // 8 waves per block -> 4 K-split pairs
#define NWAVES (NBLK * WPB)  // 640 waves == max jobs per phase

typedef __bf16 bf16_t;
typedef __attribute__((ext_vector_type(16))) __bf16 v16bf;
typedef __attribute__((ext_vector_type(8)))  __bf16 v8bf;
typedef __attribute__((ext_vector_type(8)))  float  v8f;

// ---------------------------------------------------------------------------
// Fragment loaders (bf16 WMMA 16x16x32, wave32 layouts per CDNA5 ISA 7.12.2)
// A (16xK row-major source): lane l holds row m=l&15; kbase=(l>>4)*8,
//   elements 0..7 = K kbase..kbase+7, elements 8..15 = K kbase+16..kbase+23.
// B (KxN = W^T, W row-major): lane l holds column n=l&15 (= W row),
//   kbase=(l>>4)*8, same K order as A.
// ---------------------------------------------------------------------------
__device__ __forceinline__ v16bf load_frag_a(const bf16_t* __restrict__ base,
                                             int ld, int lane) {
  const bf16_t* p = base + (size_t)(lane & 15) * ld + ((lane >> 4) << 3);
  v8bf lo = *(const v8bf*)(p);
  v8bf hi = *(const v8bf*)(p + 16);
  v16bf r;
#pragma unroll
  for (int i = 0; i < 8; ++i) { r[i] = lo[i]; r[i + 8] = hi[i]; }
  return r;
}

__device__ __forceinline__ v16bf load_frag_b(const bf16_t* __restrict__ base,
                                             int ld, int lane) {
  const bf16_t* p = base + (size_t)(lane & 15) * ld + ((lane >> 4) << 3);
  v8bf lo = *(const v8bf*)(p);
  v8bf hi = *(const v8bf*)(p + 16);
  v16bf r;
#pragma unroll
  for (int i = 0; i < 8; ++i) { r[i] = lo[i]; r[i + 8] = hi[i]; }
  return r;
}

// 64x16 strip: 4 independent accumulators (M-tiles 0..3), B fragment loaded
// once per K-slab and reused 4x -> 4-way ILP on the XDL pipe.
__device__ __forceinline__ void mma_strip4(v8f acc[4],
                                           const bf16_t* __restrict__ Abase,
                                           int lda,
                                           const bf16_t* __restrict__ Bbase,
                                           int ldb, int K, int lane) {
#pragma unroll 2
  for (int k = 0; k < K; k += 32) {
    __builtin_prefetch(Bbase + k + 256, 0, 0);  // next weight slab (L2)
    v16bf b = load_frag_b(Bbase + k, ldb, lane);
#pragma unroll
    for (int r = 0; r < 4; ++r) {
      v16bf a = load_frag_a(Abase + (size_t)(r * 16) * lda + k, lda, lane);
      acc[r] = __builtin_amdgcn_wmma_f32_16x16x32_bf16(
          false, a, false, b, (short)0, acc[r], false, false);
    }
  }
}

// C/D layout: lane l -> column n=l&15, VGPR r -> row r + (l>=16 ? 8 : 0).
__device__ __forceinline__ void store_tile_f32(float* __restrict__ base, int ld,
                                               const v8f& acc, int lane) {
  int col = lane & 15;
  int rbase = (lane >> 4) << 3;
#pragma unroll
  for (int r = 0; r < 8; ++r) base[(size_t)(rbase + r) * ld + col] = acc[r];
}

// ---------------------------------------------------------------------------
// Grid-wide barrier (count + generation in workspace)
// ---------------------------------------------------------------------------
__device__ __forceinline__ void grid_sync(unsigned* cnt, unsigned* gen) {
  __syncthreads();
  if (threadIdx.x == 0) {
    __threadfence();
    unsigned g = atomicAdd(gen, 0u);
    if (atomicAdd(cnt, 1u) == (unsigned)(NBLK - 1)) {
      atomicExch(cnt, 0u);
      __threadfence();
      atomicAdd(gen, 1u);
    } else {
      while (atomicAdd(gen, 0u) == g) __builtin_amdgcn_s_sleep(1);
    }
    __threadfence();
  }
  __syncthreads();
}

__device__ __forceinline__ float sigmoidf_(float x) {
  return 1.0f / (1.0f + __expf(-x));
}

// ---------------------------------------------------------------------------
// Helper kernels
// ---------------------------------------------------------------------------
__global__ void cvt_f32_bf16(const float* __restrict__ src,
                             bf16_t* __restrict__ dst, int n) {
  int i = blockIdx.x * blockDim.x + threadIdx.x;
  if (i < n) dst[i] = (bf16_t)src[i];
}

__global__ void init_state(const float* __restrict__ h0,
                           const float* __restrict__ c0,
                           float* hf, float* cf, bf16_t* hbf,
                           unsigned* bar, int n) {
  int i = blockIdx.x * blockDim.x + threadIdx.x;
  if (i < n) {
    float h = h0[i];
    hf[i] = h;
    cf[i] = c0[i];
    hbf[i] = (bf16_t)h;
  }
  if (i < 8) bar[i] = 0u;
}

// ---------------------------------------------------------------------------
// Persistent LSTM kernel.
// Per step t (GEMM phase): 512 gate jobs (256 column strips x 2 K-halves:
//   half0 = bias + x(t)W_x^T, half1 = h(t-1)W_h^T) fused with 128 projection
//   jobs (64 strips x 2 K-halves of p(t-1)=h(t-1)W_o^T). Wave pairs (2i,2i+1)
//   in the same block reduce partial accumulators through LDS.
// Gate phase: elementwise sigmoid/tanh -> c(t), h(t) fp32 + bf16 shadow.
// ---------------------------------------------------------------------------
__global__ void __launch_bounds__(TPB)
lstm_persistent(const bf16_t* __restrict__ xbf,    // [SEQ*BB, NIN]
                const bf16_t* __restrict__ wxbf,   // [G4, NIN]
                const bf16_t* __restrict__ whbf,   // [G4, NHID]
                const bf16_t* __restrict__ wobf,   // [NHID, NHID]
                const float*  __restrict__ bias,   // [G4]
                float* hf, float* cf, bf16_t* hbf, // [BB*NHID]
                float* aws,                        // [BB, G4]
                float* pred,                       // [SEQ*BB, NHID]
                float* hout, float* cout,          // [BB*NHID]
                unsigned* bar) {
  __shared__ float lds[(WPB / 2) * 4 * 256];       // 4 pairs x 4 tiles x 256 = 16 KB

  const int lane    = threadIdx.x & 31;
  const int winblk  = threadIdx.x >> 5;            // 0..7
  const int pairId  = winblk >> 1;                 // 0..3
  const int wave    = blockIdx.x * WPB + winblk;   // 0..639
  const int tid     = blockIdx.x * TPB + threadIdx.x;
  const int nthr    = NBLK * TPB;
  unsigned* cnt = bar;
  unsigned* gen = bar + 1;

  for (int t = 0; t <= SEQ; ++t) {
    const int aJobs = (t < SEQ) ? (2 * (G4 / 16)) : 0;   // 512 gate jobs
    const int pJobs = (t > 0) ? (2 * (NHID / 16)) : 0;   // 128 proj jobs
    const int total = aJobs + pJobs;
    const bool have = (wave < total);                    // wave-uniform
    const int job   = wave;
    const bool isA  = have && (job < aJobs);
    const int kh    = job & 1;                           // K-half within pair

    v8f acc[4];
    float* dst = nullptr;                                // even-wave store base
    int dstld = 0;

    if (isA) {
      const int nt = job >> 1;                           // 0..255 gate column strip
      if (kh == 0) {
        // half 0: bias + x(t) @ W_x^T  (K = NIN)
        const float bv = bias[nt * 16 + (lane & 15)];
#pragma unroll
        for (int r = 0; r < 4; ++r)
#pragma unroll
          for (int e = 0; e < 8; ++e) acc[r][e] = bv;
        mma_strip4(acc, xbf + (size_t)t * BB * NIN, NIN,
                   wxbf + (size_t)(nt * 16) * NIN, NIN, NIN, lane);
      } else {
        // half 1: h(t-1) @ W_h^T  (K = NHID)
#pragma unroll
        for (int r = 0; r < 4; ++r)
#pragma unroll
          for (int e = 0; e < 8; ++e) acc[r][e] = 0.0f;
        mma_strip4(acc, hbf, NHID,
                   whbf + (size_t)(nt * 16) * NHID, NHID, NHID, lane);
      }
      dst = aws + nt * 16;
      dstld = G4;
    } else if (have) {
      const int pj = job - aJobs;
      const int nt = pj >> 1;                            // 0..63 output strip
#pragma unroll
      for (int r = 0; r < 4; ++r)
#pragma unroll
        for (int e = 0; e < 8; ++e) acc[r][e] = 0.0f;
      // p(t-1) = h(t-1) @ W_o^T, K-half kh*512
      mma_strip4(acc, hbf + kh * (NHID / 2), NHID,
                 wobf + (size_t)(nt * 16) * NHID + kh * (NHID / 2), NHID,
                 NHID / 2, lane);
      dst = pred + (size_t)(t - 1) * BB * NHID + nt * 16;
      dstld = NHID;
    }

    // Pairwise K-reduction through LDS: odd wave publishes, even wave sums.
    if (have && kh == 1) {
      const int base = pairId * 1024;
#pragma unroll
      for (int r = 0; r < 4; ++r)
#pragma unroll
        for (int e = 0; e < 8; ++e)
          lds[base + r * 256 + lane * 8 + e] = acc[r][e];
    }
    __syncthreads();
    if (have && kh == 0) {
      const int base = pairId * 1024;
#pragma unroll
      for (int r = 0; r < 4; ++r) {
#pragma unroll
        for (int e = 0; e < 8; ++e)
          acc[r][e] += lds[base + r * 256 + lane * 8 + e];
        store_tile_f32(dst + (size_t)(r * 16) * dstld, dstld, acc[r], lane);
      }
    }
    grid_sync(cnt, gen);

    if (t < SEQ) {
      for (int idx = tid; idx < BB * NHID; idx += nthr) {
        const int b = idx >> 10;
        const int j = idx & (NHID - 1);
        const float ai = aws[(size_t)b * G4 + j];
        const float af = aws[(size_t)b * G4 + NHID + j];
        const float ao = aws[(size_t)b * G4 + 2 * NHID + j];
        const float ag = aws[(size_t)b * G4 + 3 * NHID + j];
        const float ig = sigmoidf_(ai);
        const float fg = sigmoidf_(af);
        const float og = sigmoidf_(ao);
        const float g  = tanhf(ag);
        const float c  = fg * cf[idx] + ig * g;
        const float h  = og * tanhf(c);
        cf[idx]  = c;
        hf[idx]  = h;
        hbf[idx] = (bf16_t)h;
      }
      grid_sync(cnt, gen);
    }
  }

  // Final h, c outputs
  for (int idx = tid; idx < BB * NHID; idx += nthr) {
    hout[idx] = hf[idx];
    cout[idx] = cf[idx];
  }
}

// ---------------------------------------------------------------------------
extern "C" void kernel_launch(void* const* d_in, const int* in_sizes, int n_in,
                              void* d_out, int out_size, void* d_ws, size_t ws_size,
                              hipStream_t stream) {
  const float* x    = (const float*)d_in[0];  // [SEQ, BB, NIN]
  const float* h0   = (const float*)d_in[1];  // [BB, NHID]
  const float* c0   = (const float*)d_in[2];  // [BB, NHID]
  const float* w_x  = (const float*)d_in[3];  // [G4, NIN]
  const float* w_h  = (const float*)d_in[4];  // [G4, NHID]
  const float* bias = (const float*)d_in[5];  // [G4]
  const float* w_o  = (const float*)d_in[6];  // [NHID, NHID]

  // Workspace carve-up (~84 MB total)
  char* ws = (char*)d_ws;
  size_t off = 0;
  auto take = [&](size_t bytes) {
    char* p = ws + off;
    off = (off + bytes + 255) & ~(size_t)255;
    return p;
  };
  unsigned* bar  = (unsigned*)take(256);
  bf16_t* xbf    = (bf16_t*)take((size_t)SEQ * BB * NIN * sizeof(bf16_t));
  bf16_t* wxbf   = (bf16_t*)take((size_t)G4 * NIN * sizeof(bf16_t));
  bf16_t* whbf   = (bf16_t*)take((size_t)G4 * NHID * sizeof(bf16_t));
  bf16_t* wobf   = (bf16_t*)take((size_t)NHID * NHID * sizeof(bf16_t));
  bf16_t* hbf    = (bf16_t*)take((size_t)BB * NHID * sizeof(bf16_t));
  float*  hf     = (float*)take((size_t)BB * NHID * sizeof(float));
  float*  cf     = (float*)take((size_t)BB * NHID * sizeof(float));
  float*  aws    = (float*)take((size_t)BB * G4 * sizeof(float));
  (void)ws_size; (void)in_sizes; (void)n_in; (void)out_size;

  float* pred = (float*)d_out;                          // [SEQ*BB*NHID]
  float* hout = pred + (size_t)SEQ * BB * NHID;
  float* cout = hout + (size_t)BB * NHID;

  const int cvtT = 256;
  {
    int n = SEQ * BB * NIN;
    cvt_f32_bf16<<<(n + cvtT - 1) / cvtT, cvtT, 0, stream>>>(x, xbf, n);
  }
  {
    int n = G4 * NIN;
    cvt_f32_bf16<<<(n + cvtT - 1) / cvtT, cvtT, 0, stream>>>(w_x, wxbf, n);
  }
  {
    int n = G4 * NHID;
    cvt_f32_bf16<<<(n + cvtT - 1) / cvtT, cvtT, 0, stream>>>(w_h, whbf, n);
  }
  {
    int n = NHID * NHID;
    cvt_f32_bf16<<<(n + cvtT - 1) / cvtT, cvtT, 0, stream>>>(w_o, wobf, n);
  }
  {
    int n = BB * NHID;
    init_state<<<(n + cvtT - 1) / cvtT, cvtT, 0, stream>>>(h0, c0, hf, cf, hbf,
                                                           bar, n);
  }
  lstm_persistent<<<NBLK, TPB, 0, stream>>>(xbf, wxbf, whbf, wobf, bias,
                                            hf, cf, hbf, aws,
                                            pred, hout, cout, bar);
}
